// siiformer_ca1_48077863911759
// MI455X (gfx1250) — matrix operand order repaired and checked by reading, not compiled, not run on hardware
//
#include <hip/hip_runtime.h>
#include <hip/hip_bf16.h>
#include <math.h>

// Problem constants (fixed by the reference)
#define NQ 16384
#define NK 7933
#define NK_PAD 7936
#define DIN 1024
#define HD 512
#define RANK_ASC 11469   // NQ - int(0.3*NQ) = 16384 - 4915

typedef __attribute__((ext_vector_type(16))) _Float16 v16h;
typedef __attribute__((ext_vector_type(8)))  float    v8f;
typedef __attribute__((ext_vector_type(4)))  float    f32x4;

// ---------------------------------------------------------------------------
// GEMM: C[M,N] = tanh(A[M,K] @ W[N,K]^T + bias[N]) using v_wmma_f32_16x16x32_f16
// One wave computes one 16x16 M-tile x TWO adjacent 16-col N-tiles, so the
// A operand (load + f32->f16 convert) is reused across 2 WMMAs.
// 4 waves (4 M-tiles) per 128-thread block. f32 accumulate.
// ---------------------------------------------------------------------------
__global__ __launch_bounds__(128)
void gemm_tanh_wmma(const float* __restrict__ A, const float* __restrict__ W,
                    const float* __restrict__ bias, float* __restrict__ C,
                    int M, int N, int K) {
  const int lane = threadIdx.x & 31;
  const int wave = threadIdx.x >> 5;
  const int mt   = blockIdx.x * 4 + wave;     // M tile index
  const int nt   = blockIdx.y * 2;            // first of two N tile indices
  const int hi   = lane >> 4;                 // 0: lanes 0-15, 1: lanes 16-31
  const int l16  = lane & 15;

  int mrow = mt * 16 + l16;
  if (mrow >= M) mrow = M - 1;                // clamp loads; stores are guarded
  const int ncol0 = nt * 16 + l16;
  const int ncol1 = ncol0 + 16;

  const float* __restrict__ arow  = A + (size_t)mrow * K;
  const float* __restrict__ wrow0 = W + (size_t)ncol0 * K;
  const float* __restrict__ wrow1 = W + (size_t)ncol1 * K;

  // A 16x32 — lanes 0-15 hold K 0..15, lanes 16-31 hold K 16..31 (contiguous, matches B)
  const int ag0 = hi ? 16 : 0;
  // B 32x16 — N = lane&15, halves h -> K = h + 16*(lane>=16)
  const int bk0 = hi ? 16 : 0;

  v8f acc0, acc1;
  const float bv0 = bias[ncol0];
  const float bv1 = bias[ncol1];
#pragma unroll
  for (int i = 0; i < 8; ++i) { acc0[i] = bv0; acc1[i] = bv1; }

#pragma unroll 2
  for (int kb = 0; kb < K; kb += 32) {
    f32x4 a0 = *(const f32x4*)(arow + kb + ag0 + 0);
    f32x4 a1 = *(const f32x4*)(arow + kb + ag0 + 4);
    f32x4 a2 = *(const f32x4*)(arow + kb + ag0 + 8);
    f32x4 a3 = *(const f32x4*)(arow + kb + ag0 + 12);
    f32x4 b0 = *(const f32x4*)(wrow0 + kb + bk0 + 0);
    f32x4 b1 = *(const f32x4*)(wrow0 + kb + bk0 + 4);
    f32x4 b2 = *(const f32x4*)(wrow0 + kb + bk0 + 8);
    f32x4 b3 = *(const f32x4*)(wrow0 + kb + bk0 + 12);
    f32x4 c0 = *(const f32x4*)(wrow1 + kb + bk0 + 0);
    f32x4 c1 = *(const f32x4*)(wrow1 + kb + bk0 + 4);
    f32x4 c2 = *(const f32x4*)(wrow1 + kb + bk0 + 8);
    f32x4 c3 = *(const f32x4*)(wrow1 + kb + bk0 + 12);

    v16h av, bw0, bw1;
#pragma unroll
    for (int j = 0; j < 4; ++j) {
      av[0  + j] = (_Float16)a0[j];
      av[4  + j] = (_Float16)a1[j];
      av[8  + j] = (_Float16)a2[j];
      av[12 + j] = (_Float16)a3[j];
      bw0[0  + j] = (_Float16)b0[j];
      bw0[4  + j] = (_Float16)b1[j];
      bw0[8  + j] = (_Float16)b2[j];
      bw0[12 + j] = (_Float16)b3[j];
      bw1[0  + j] = (_Float16)c0[j];
      bw1[4  + j] = (_Float16)c1[j];
      bw1[8  + j] = (_Float16)c2[j];
      bw1[12 + j] = (_Float16)c3[j];
    }
    acc0 = __builtin_amdgcn_wmma_f32_16x16x32_f16(
        false, av, false, bw0, (short)0, acc0, false, false);
    acc1 = __builtin_amdgcn_wmma_f32_16x16x32_f16(
        false, av, false, bw1, (short)0, acc1, false, false);
  }

  // C/D layout: VGPR i, lanes 0-15 -> M=i, lanes 16-31 -> M=i+8; N = lane&15
#pragma unroll
  for (int i = 0; i < 8; ++i) {
    const int r = mt * 16 + i + (hi ? 8 : 0);
    if (r < M) {
      C[(size_t)r * N + ncol0] = tanhf(acc0[i]);
      C[(size_t)r * N + ncol1] = tanhf(acc1[i]);
    }
  }
}

// ---------------------------------------------------------------------------
// sk[k] = wa_w[k] / max(||kh[k,:]||, 1e-12)  — one wave per row
// ---------------------------------------------------------------------------
__global__ __launch_bounds__(256)
void k_scale_kernel(const float* __restrict__ kh, const float* __restrict__ wa_w,
                    float* __restrict__ sk) {
  const int lane = threadIdx.x & 31;
  const int wave = threadIdx.x >> 5;
  const int row  = blockIdx.x * 8 + wave;
  if (row >= NK) return;
  const float* r = kh + (size_t)row * HD;
  float ss = 0.f;
  for (int j = lane; j < HD; j += 32) { float v = r[j]; ss += v * v; }
#pragma unroll
  for (int off = 16; off; off >>= 1) ss += __shfl_xor(ss, off, 32);
  if (lane == 0) sk[row] = wa_w[row] / fmaxf(sqrtf(ss), 1e-12f);
}

// ---------------------------------------------------------------------------
// kw[c] += sum_k sk[k] * kh[k,c]  (split-K, atomic accumulate)
// grid (2 col-chunks of 256, 31 k-chunks of 256)
// ---------------------------------------------------------------------------
__global__ __launch_bounds__(256)
void kw_kernel(const float* __restrict__ kh, const float* __restrict__ sk,
               float* __restrict__ kw) {
  const int c  = blockIdx.x * 256 + threadIdx.x;
  const int k0 = blockIdx.y * 256;
  const int k1 = (k0 + 256 < NK) ? (k0 + 256) : NK;
  float acc = 0.f;
  for (int k = k0; k < k1; ++k) acc += sk[k] * kh[(size_t)k * HD + c];
  atomicAdd(&kw[c], acc);
}

// ---------------------------------------------------------------------------
// A1[q] = (qh[q,:]·kw) / max(||qh[q,:]||,1e-12) + wa_b  — one wave per row
// ---------------------------------------------------------------------------
__global__ __launch_bounds__(256)
void a1_kernel(const float* __restrict__ qh, const float* __restrict__ kw,
               const float* __restrict__ wa_b, float* __restrict__ A1) {
  const int lane = threadIdx.x & 31;
  const int wave = threadIdx.x >> 5;
  const int row  = blockIdx.x * 8 + wave;
  if (row >= NQ) return;
  const float* r = qh + (size_t)row * HD;
  float ss = 0.f, dt = 0.f;
  for (int j = lane; j < HD; j += 32) {
    float v = r[j];
    ss += v * v;
    dt += v * kw[j];
  }
#pragma unroll
  for (int off = 16; off; off >>= 1) {
    ss += __shfl_xor(ss, off, 32);
    dt += __shfl_xor(dt, off, 32);
  }
  if (lane == 0) A1[row] = dt / fmaxf(sqrtf(ss), 1e-12f) + wa_b[0];
}

// ---------------------------------------------------------------------------
// Exact order statistic: ascending 0-based rank RANK_ASC via 4-pass MSB radix
// select over order-preserving uint keys. Single block.
// ---------------------------------------------------------------------------
__global__ __launch_bounds__(256)
void select_kernel(const float* __restrict__ A1, float* __restrict__ thre) {
  __shared__ unsigned hist[256];
  __shared__ unsigned s_prefix;
  __shared__ int s_remaining;
  if (threadIdx.x == 0) { s_prefix = 0u; s_remaining = RANK_ASC; }
  for (int pass = 3; pass >= 0; --pass) {
    for (int b = threadIdx.x; b < 256; b += blockDim.x) hist[b] = 0u;
    __syncthreads();
    const unsigned prefix = s_prefix;
    const unsigned mask = (pass == 3) ? 0u : (0xFFFFFFFFu << ((pass + 1) * 8));
    for (int i = threadIdx.x; i < NQ; i += blockDim.x) {
      unsigned bits = __float_as_uint(A1[i]);
      unsigned key  = (bits & 0x80000000u) ? ~bits : (bits | 0x80000000u);
      if ((key & mask) == prefix)
        atomicAdd(&hist[(key >> (pass * 8)) & 255u], 1u);
    }
    __syncthreads();
    if (threadIdx.x == 0) {
      int rem = s_remaining;
      unsigned cum = 0; int b = 0;
      for (; b < 256; ++b) {
        unsigned c = hist[b];
        if (cum + c > (unsigned)rem) break;
        cum += c;
      }
      s_prefix = prefix | ((unsigned)b << (pass * 8));
      s_remaining = rem - (int)cum;
    }
    __syncthreads();
  }
  if (threadIdx.x == 0) {
    unsigned key  = s_prefix;
    unsigned bits = (key & 0x80000000u) ? (key & 0x7FFFFFFFu) : ~key;
    *thre = __uint_as_float(bits);
  }
}

// ---------------------------------------------------------------------------
// attn = softmax(threshold(A1, thre, 0)) over all NQ. Single 1024-thread block.
// ---------------------------------------------------------------------------
__global__ __launch_bounds__(1024)
void softmax_kernel(const float* __restrict__ A1, const float* __restrict__ thre_p,
                    float* __restrict__ attn) {
  __shared__ float red[32];
  const float thre = *thre_p;
  const int lane = threadIdx.x & 31;
  const int wv   = threadIdx.x >> 5;
  const int nw   = blockDim.x >> 5;

  float lmax = -3.4e38f;
  for (int i = threadIdx.x; i < NQ; i += blockDim.x) {
    float v = A1[i]; v = (v > thre) ? v : 0.f;
    lmax = fmaxf(lmax, v);
  }
#pragma unroll
  for (int off = 16; off; off >>= 1) lmax = fmaxf(lmax, __shfl_xor(lmax, off, 32));
  if (lane == 0) red[wv] = lmax;
  __syncthreads();
  if (threadIdx.x == 0) {
    float m = red[0];
    for (int i = 1; i < nw; ++i) m = fmaxf(m, red[i]);
    red[0] = m;
  }
  __syncthreads();
  const float m = red[0];
  __syncthreads();

  float lsum = 0.f;
  for (int i = threadIdx.x; i < NQ; i += blockDim.x) {
    float v = A1[i]; v = (v > thre) ? v : 0.f;
    lsum += expf(v - m);
  }
#pragma unroll
  for (int off = 16; off; off >>= 1) lsum += __shfl_xor(lsum, off, 32);
  if (lane == 0) red[wv] = lsum;
  __syncthreads();
  if (threadIdx.x == 0) {
    float s = red[0];
    for (int i = 1; i < nw; ++i) s += red[i];
    red[0] = s;
  }
  __syncthreads();
  const float inv = 1.f / red[0];

  for (int i = threadIdx.x; i < NQ; i += blockDim.x) {
    float v = A1[i]; v = (v > thre) ? v : 0.f;
    attn[i] = expf(v - m) * inv;
  }
}

// ---------------------------------------------------------------------------
// zacc[c] += sum_q attn[q] * qh[q,c]   (split over q, atomic accumulate)
// grid (2 col-chunks of 256, 32 q-chunks of 512)
// ---------------------------------------------------------------------------
__global__ __launch_bounds__(256)
void z_kernel(const float* __restrict__ qh, const float* __restrict__ attn,
              float* __restrict__ zacc) {
  const int c  = blockIdx.x * 256 + threadIdx.x;
  const int q0 = blockIdx.y * 512;
  float acc = 0.f;
  for (int q = q0; q < q0 + 512; ++q) acc += attn[q] * qh[(size_t)q * HD + c];
  atomicAdd(&zacc[c], acc);
}

// ---------------------------------------------------------------------------
// y = zacc · cls_w + cls_b   (single block)
// ---------------------------------------------------------------------------
__global__ __launch_bounds__(256)
void y_kernel(const float* __restrict__ zacc, const float* __restrict__ cls_w,
              const float* __restrict__ cls_b, float* __restrict__ y) {
  __shared__ float red[8];
  const int lane = threadIdx.x & 31;
  const int wv   = threadIdx.x >> 5;
  float p = 0.f;
  for (int c = threadIdx.x; c < HD; c += blockDim.x) p += zacc[c] * cls_w[c];
#pragma unroll
  for (int off = 16; off; off >>= 1) p += __shfl_xor(p, off, 32);
  if (lane == 0) red[wv] = p;
  __syncthreads();
  if (threadIdx.x == 0) {
    float s = 0.f;
    for (int i = 0; i < 8; ++i) s += red[i];
    y[0] = s + cls_b[0];
  }
}

// ---------------------------------------------------------------------------
extern "C" void kernel_launch(void* const* d_in, const int* in_sizes, int n_in,
                              void* d_out, int out_size, void* d_ws, size_t ws_size,
                              hipStream_t stream) {
  (void)in_sizes; (void)n_in; (void)out_size; (void)ws_size;
  const float* query = (const float*)d_in[0];
  const float* key_x = (const float*)d_in[1];
  const float* wq_w  = (const float*)d_in[2];
  const float* wq_b  = (const float*)d_in[3];
  const float* wk_w  = (const float*)d_in[4];
  const float* wk_b  = (const float*)d_in[5];
  const float* wa_w  = (const float*)d_in[6];
  const float* wa_b  = (const float*)d_in[7];
  const float* cls_w = (const float*)d_in[8];
  const float* cls_b = (const float*)d_in[9];

  float* out = (float*)d_out;           // out[0] = y_prob; out[1..NQ] = attn_t
  float* ws  = (float*)d_ws;

  float* qh   = ws;                                 // NQ*HD
  float* kh   = qh + (size_t)NQ * HD;               // NK_PAD*HD
  float* sk   = kh + (size_t)NK_PAD * HD;           // NK_PAD
  float* A1   = sk + NK_PAD;                        // NQ
  float* kw   = A1 + NQ;                            // HD
  float* zacc = kw + HD;                            // HD
  float* thre = zacc + HD;                          // 1

  // zero the atomic accumulators (kw and zacc are contiguous)
  hipMemsetAsync(kw, 0, (size_t)(2 * HD) * sizeof(float), stream);

  // GEMM1: q = tanh(query @ wq_w^T): 1024 M-tiles / 4 waves, 16 N-pair tiles
  gemm_tanh_wmma<<<dim3(NQ / 64, HD / 32), dim3(128), 0, stream>>>(
      query, wq_w, wq_b, qh, NQ, HD, DIN);
  // GEMM2: k = tanh(key_x @ wk_w^T): 496 M-tiles / 4 waves = 124 blocks
  gemm_tanh_wmma<<<dim3(NK_PAD / 64, HD / 32), dim3(128), 0, stream>>>(
      key_x, wk_w, wk_b, kh, NK, HD, DIN);

  k_scale_kernel<<<dim3(NK_PAD / 8), dim3(256), 0, stream>>>(kh, wa_w, sk);
  kw_kernel<<<dim3(2, NK_PAD / 256), dim3(256), 0, stream>>>(kh, sk, kw);
  a1_kernel<<<dim3(NQ / 8), dim3(256), 0, stream>>>(qh, kw, wa_b, A1);
  select_kernel<<<dim3(1), dim3(256), 0, stream>>>(A1, thre);
  softmax_kernel<<<dim3(1), dim3(1024), 0, stream>>>(A1, thre, out + 1);
  z_kernel<<<dim3(2, NQ / 512), dim3(256), 0, stream>>>(qh, out + 1, zacc);
  y_kernel<<<dim3(1), dim3(256), 0, stream>>>(zacc, cls_w, cls_b, out);
}
